// GPT_5832565588418
// MI455X (gfx1250) — matrix operand order repaired and checked by reading, not compiled, not run on hardware
//
#include <hip/hip_runtime.h>
#include <cstdint>

// ---------------------------------------------------------------------------
// GPT forward pass for gfx1250 (MI455X).
// - All weights converted once to f16 in workspace (~179MB -> L2-resident).
// - GEMMs: 128x128 block tile, 8 waves, 32x64 per wave, 8x v_wmma_f32_16x16x32_f16
//   per 32-deep K step.
// - A tile staged to LDS by the Tensor Data Mover (tensor_load_to_lds, wave 0,
//   TENSORcnt) in the FULL variant; by global_load_async_to_lds_b128 (ASYNCcnt)
//   in the guarded variant. B tile staged transposed via b128 loads + ds stores.
// ---------------------------------------------------------------------------

typedef _Float16 v16h __attribute__((ext_vector_type(16)));
typedef _Float16 h8   __attribute__((ext_vector_type(8)));
typedef float    v8f  __attribute__((ext_vector_type(8)));
typedef unsigned v4u  __attribute__((ext_vector_type(4)));
typedef unsigned v8u  __attribute__((ext_vector_type(8)));

#define BM 128
#define BN 128
#define BK 32

static __device__ __forceinline__ unsigned lds_off(const void* p) {
  return (unsigned)(uintptr_t)p;   // low 32 bits of flat LDS address = LDS byte offset
}

static __device__ __forceinline__ void async_load_b128(unsigned lds_addr, const void* gaddr) {
  asm volatile("global_load_async_to_lds_b128 %0, %1, off"
               :: "v"(lds_addr), "v"(gaddr) : "memory");
}
static __device__ __forceinline__ void wait_async0() {
  asm volatile("s_wait_asynccnt 0x0" ::: "memory");
}

// TDM: load a 2D tile (tile_dim0 x tile_dim1, f16) of a row-major tensor into
// contiguous LDS. Issued once by the executing wave; tracked with TENSORcnt.
static __device__ __forceinline__ void tdm_load_tile_f16(
    unsigned lds_addr, const void* gptr, unsigned tensor_d0, unsigned tensor_d1,
    unsigned stride0, unsigned tile_d0, unsigned tile_d1)
{
  unsigned long long ga = (unsigned long long)(uintptr_t)gptr;
  v4u g0;
  g0[0] = 1u;                                            // count=1, user mode
  g0[1] = lds_addr;                                      // lds_addr
  g0[2] = (unsigned)(ga & 0xFFFFFFFFu);                  // global_addr[31:0]
  g0[3] = (unsigned)((ga >> 32) & 0x01FFFFFFu) | (2u << 30); // addr[56:32] | type=2
  v8u g1;
  g1[0] = 0x00010000u;                                   // data_size=1 (2 bytes)
  g1[1] = (tensor_d0 & 0xFFFFu) << 16;                   // tensor_dim0[15:0]
  g1[2] = (tensor_d0 >> 16) | ((tensor_d1 & 0xFFFFu) << 16); // dim0 hi | dim1 lo
  g1[3] = (tensor_d1 >> 16) | (tile_d0 << 16);           // dim1 hi | tile_dim0
  g1[4] = tile_d1;                                       // tile_dim1 | tile_dim2=0
  g1[5] = stride0;                                       // tensor_dim0_stride lo32
  g1[6] = 0u;                                            // stride0 hi | stride1 lo
  g1[7] = 0u;
  v4u g2 = {0u, 0u, 0u, 0u};                             // dim2/3 unused (tile_dim3=0)
  v4u g3 = {0u, 0u, 0u, 0u};
  asm volatile("tensor_load_to_lds %0, %1, %2, %3"
               :: "s"(g0), "s"(g1), "s"(g2), "s"(g3) : "memory");
}

static __device__ __forceinline__ v8f wmma16(v16h a, v16h b, v8f c) {
  return __builtin_amdgcn_wmma_f32_16x16x32_f16(false, a, false, b, (short)0, c, false, false);
}

// A fragment 16x32 f16: lane L -> row = L%16; lanes 0-15 hold K 0..7 & 16..23,
// lanes 16-31 hold K 8..15 & 24..31 (ISA 16-bit A layout).
static __device__ __forceinline__ v16h load_a_frag(const _Float16* base, int lane) {
  int r = lane & 15, sel = lane >> 4;
  const _Float16* p = base + r * BK + sel * 8;
  h8 lo = *(const h8*)p;
  h8 hi = *(const h8*)(p + 16);
  v16h o;
#pragma unroll
  for (int i = 0; i < 8; ++i) { o[i] = lo[i]; o[8 + i] = hi[i]; }
  return o;
}

// B fragment 32x16 f16 from transposed LDS tile Bs[BN][BK]:
// lane L -> col = L%16, K = (L/16)*8 + 0..7 & 16 + (L/16)*8 + 0..7.
static __device__ __forceinline__ v16h load_b_frag(const _Float16* base, int lane) {
  int c = lane & 15, sel = lane >> 4;
  const _Float16* p = base + c * BK + sel * 8;
  h8 lo = *(const h8*)p;
  h8 hi = *(const h8*)(p + 16);
  v16h o;
#pragma unroll
  for (int i = 0; i < 8; ++i) { o[i] = lo[i]; o[8 + i] = hi[i]; }
  return o;
}

// C[M,N] = A[M,K](f16) * B[K,N](f16) + bias[N] (+ resid[M,N]); C f32.
// M % 128 == 0, K % 32 == 0 always; FULL => N % 128 == 0 (no guards anywhere).
template <bool FULL>
__global__ __launch_bounds__(256)
void k_gemm_f16(const _Float16* __restrict__ A, const _Float16* __restrict__ Bw,
                const float* __restrict__ bias, const float* __restrict__ resid,
                float* __restrict__ C, int M, int N, int K)
{
  __shared__ alignas(16) _Float16 As[BM][BK];   // 8 KiB
  __shared__ alignas(16) _Float16 Bs[BN][BK];   // 8 KiB, transposed tile
  (void)M;

  const int tid  = threadIdx.x;
  const int lane = tid & 31;
  const int wave = tid >> 5;          // 0..7
  const int wm   = wave >> 1;         // 0..3 (M dir, 32 rows each)
  const int wn   = wave & 1;          // 0..1 (N dir, 64 cols each)

  const int m0 = blockIdx.y * BM;
  const int n0 = blockIdx.x * BN;

  v8f acc[2][4] = {};

  // per-thread async A map (guarded variant): two b128 per thread = 128x32 tile
  const int ar = tid >> 2;            // 0..63
  const int ac = (tid & 3) * 8;       // 0,8,16,24
  const unsigned a_lds0 = lds_off(&As[ar][ac]);
  const unsigned a_lds1 = lds_off(&As[ar + 64][ac]);
  const unsigned as_base = lds_off(&As[0][0]);

  for (int k0 = 0; k0 < K; k0 += BK) {
    // ---- stage A tile ----
    if (FULL) {
      if (wave == 0)
        tdm_load_tile_f16(as_base, A + (size_t)m0 * K + k0,
                          (unsigned)K, (unsigned)M, (unsigned)K, BK, BM);
    } else {
      async_load_b128(a_lds0, A + (size_t)(m0 + ar) * K + (k0 + ac));
      async_load_b128(a_lds1, A + (size_t)(m0 + ar + 64) * K + (k0 + ac));
    }

    // ---- stage B tile transposed: 32x128 halves, 2 groups of 8 per thread ----
#pragma unroll
    for (int gg = 0; gg < 2; ++gg) {
      int g    = tid + gg * 256;
      int row  = g >> 4;            // 0..31  (k within tile)
      int ct   = (g & 15) * 8;      // 0..120 (n within tile)
      int gcol = n0 + ct;
      const _Float16* src = Bw + (size_t)(k0 + row) * N + gcol;
      if (FULL) {
        if (k0 + BK < K)
          __builtin_prefetch(Bw + (size_t)(k0 + BK + row) * N + gcol, 0, 1);
        h8 dv = *(const h8*)src;
#pragma unroll
        for (int i = 0; i < 8; ++i) Bs[ct + i][row] = dv[i];
      } else {
        // branch-free clamped loads (N=64 path only)
        const _Float16* rowp = Bw + (size_t)(k0 + row) * N;
#pragma unroll
        for (int i = 0; i < 8; ++i) {
          bool in = (gcol + i) < N;
          _Float16 v = rowp[in ? (gcol + i) : 0];
          Bs[ct + i][row] = in ? v : (_Float16)0;
        }
      }
    }

    if (FULL) {
      if (wave == 0) __builtin_amdgcn_s_wait_tensorcnt(0);
    } else {
      wait_async0();
    }
    __syncthreads();

    // ---- compute: 2 A frags x 4 B frags = 8 WMMA ----
    v16h af[2];
#pragma unroll
    for (int i = 0; i < 2; ++i)
      af[i] = load_a_frag(&As[wm * 32 + i * 16][0], lane);
#pragma unroll
    for (int j = 0; j < 4; ++j) {
      v16h bf = load_b_frag(&Bs[wn * 64 + j * 16][0], lane);
      acc[0][j] = wmma16(af[0], bf, acc[0][j]);
      acc[1][j] = wmma16(af[1], bf, acc[1][j]);
    }
    __syncthreads();
  }

  // epilogue: lanes 0-15 VGPR e -> M=e; lanes 16-31 -> M=8+e; N=lane%16
#pragma unroll
  for (int i = 0; i < 2; ++i) {
#pragma unroll
    for (int j = 0; j < 4; ++j) {
      int col = n0 + wn * 64 + j * 16 + (lane & 15);
      if (!FULL && col >= N) continue;
      int rbase = m0 + wm * 32 + i * 16 + (lane >> 4) * 8;
#pragma unroll
      for (int e = 0; e < 8; ++e) {
        size_t idx = (size_t)(rbase + e) * N + col;
        float v = acc[i][j][e];
        if (bias)  v += bias[col];
        if (resid) v += resid[idx];
        C[idx] = v;
      }
    }
  }
}

// x[bt,d] = tok_emb[tokens[bt], d] + pos_emb[t, d]
__global__ void k_embed(const int* __restrict__ tokens, const float* __restrict__ te,
                        const float* __restrict__ pe, float* __restrict__ x)
{
  size_t i = (size_t)blockIdx.x * 256 + threadIdx.x;
  if (i >= (size_t)2048 * 1024) return;
  int bt = (int)(i >> 10), d = (int)(i & 1023);
  int t  = bt & 1023;
  int tok = tokens[bt];
  x[i] = te[(size_t)tok * 1024 + d] + pe[(size_t)t * 1024 + d];
}

__global__ void k_cvt_f32_f16(const float* __restrict__ s, _Float16* __restrict__ d, size_t n)
{
  size_t i = (size_t)blockIdx.x * 256 + threadIdx.x;
  if (i < n) d[i] = (_Float16)s[i];
}

// one row per block; f32 in, f16 out
__global__ __launch_bounds__(256)
void k_layernorm(const float* __restrict__ x, const float* __restrict__ w,
                 const float* __restrict__ b, _Float16* __restrict__ out, int D)
{
  __shared__ float r1[256], r2[256];
  int row = blockIdx.x, tid = threadIdx.x;
  const float* xr = x + (size_t)row * D;
  float s = 0.f, s2 = 0.f;
  for (int i = tid; i < D; i += 256) { float v = xr[i]; s += v; s2 += v * v; }
  r1[tid] = s; r2[tid] = s2; __syncthreads();
  for (int off = 128; off > 0; off >>= 1) {
    if (tid < off) { r1[tid] += r1[tid + off]; r2[tid] += r2[tid + off]; }
    __syncthreads();
  }
  float mean = r1[0] / D;
  float var  = r2[0] / D - mean * mean;
  float rstd = rsqrtf(var + 1e-5f);
  for (int i = tid; i < D; i += 256)
    out[(size_t)row * D + i] = (_Float16)((xr[i] - mean) * rstd * w[i] + b[i]);
}

// RoPE on q laid out [B*T, H*64]; rotate first 32 dims of each head, pos = t.
__global__ void k_rope_q(float* __restrict__ q)
{
  size_t idx = (size_t)blockIdx.x * 256 + threadIdx.x;
  if (idx >= (size_t)2048 * 16 * 16) return;
  int i  = (int)(idx & 15);
  int h  = (int)((idx >> 4) & 15);
  int bt = (int)(idx >> 8);
  float pos = (float)(bt & 1023);
  float* p = q + (size_t)bt * 1024 + h * 64;
  const float ln1e4 = 9.210340371976184f;
  float tha = __expf(-((float)i      / 32.f) * ln1e4);
  float thb = __expf(-((float)(i+16) / 32.f) * ln1e4);
  float fa = pos * tha, fb = pos * thb;
  float a = p[i], c = p[i + 16];
  p[i]      = a * __cosf(fa) - c * __sinf(fa);
  p[i + 16] = c * __cosf(fb) + a * __sinf(fb);
}

// RoPE on k laid out [B*T, 64]
__global__ void k_rope_k(float* __restrict__ k)
{
  size_t idx = (size_t)blockIdx.x * 256 + threadIdx.x;
  if (idx >= (size_t)2048 * 16) return;
  int i  = (int)(idx & 15);
  int bt = (int)(idx >> 4);
  float pos = (float)(bt & 1023);
  float* p = k + (size_t)bt * 64;
  const float ln1e4 = 9.210340371976184f;
  float tha = __expf(-((float)i      / 32.f) * ln1e4);
  float thb = __expf(-((float)(i+16) / 32.f) * ln1e4);
  float fa = pos * tha, fb = pos * thb;
  float a = p[i], c = p[i + 16];
  p[i]      = a * __cosf(fa) - c * __sinf(fa);
  p[i + 16] = c * __cosf(fb) + a * __sinf(fb);
}

// one block per (b,h,t): causal softmax attention (MQA: shared k/v per batch)
__global__ __launch_bounds__(256)
void k_attn(const float* __restrict__ qf, const float* __restrict__ kf,
            const float* __restrict__ vf, _Float16* __restrict__ out)
{
  const int T_ = 1024, H_ = 16;
  int tid = threadIdx.x;
  int t = blockIdx.x % T_;
  int h = (blockIdx.x / T_) % H_;
  int b = blockIdx.x / (T_ * H_);
  __shared__ float qs[64];
  __shared__ float sc[1024];
  __shared__ float red[256];
  __shared__ float oacc[4][64];
  const size_t bt = (size_t)b * T_ + t;
  const float* qrow = qf + bt * 1024 + (size_t)h * 64;
  if (tid < 64) qs[tid] = qrow[tid];
  __syncthreads();

  float lmax = -3.0e38f;
  for (int j = tid; j <= t; j += 256) {
    const float* kr = kf + ((size_t)b * T_ + j) * 64;
    float s = 0.f;
#pragma unroll 8
    for (int d = 0; d < 64; ++d) s += qs[d] * kr[d];
    s *= 0.125f;
    sc[j] = s;
    lmax = fmaxf(lmax, s);
  }
  red[tid] = lmax; __syncthreads();
  for (int off = 128; off > 0; off >>= 1) {
    if (tid < off) red[tid] = fmaxf(red[tid], red[tid + off]);
    __syncthreads();
  }
  float mx = red[0]; __syncthreads();

  float lsum = 0.f;
  for (int j = tid; j <= t; j += 256) { float e = __expf(sc[j] - mx); sc[j] = e; lsum += e; }
  red[tid] = lsum; __syncthreads();
  for (int off = 128; off > 0; off >>= 1) {
    if (tid < off) red[tid] += red[tid + off];
    __syncthreads();
  }
  float inv = 1.0f / red[0];

  int d = tid & 63, part = tid >> 6;
  float a = 0.f;
  for (int j = part; j <= t; j += 4) a += sc[j] * vf[((size_t)b * T_ + j) * 64 + d];
  oacc[part][d] = a; __syncthreads();
  if (tid < 64) {
    float o = (oacc[0][tid] + oacc[1][tid] + oacc[2][tid] + oacc[3][tid]) * inv;
    out[bt * 1024 + (size_t)h * 64 + tid] = (_Float16)o;
  }
}

__global__ void k_silu_gate(const float* __restrict__ u, const float* __restrict__ t,
                            _Float16* __restrict__ g, size_t n)
{
  size_t i = (size_t)blockIdx.x * 256 + threadIdx.x;
  if (i < n) {
    float a = u[i];
    float s = a / (1.f + __expf(-a));
    g[i] = (_Float16)(s * t[i]);
  }
}

// ---------------------------------------------------------------------------

extern "C" void kernel_launch(void* const* d_in, const int* in_sizes, int n_in,
                              void* d_out, int out_size, void* d_ws, size_t ws_size,
                              hipStream_t stream) {
  (void)in_sizes; (void)n_in; (void)out_size; (void)ws_size;
  const int L = 4, D = 1024, H = 16, HD = 64, FF = 4096, V = 32000, B = 2, T = 1024;
  const int M = B * T;

  const int*   tokens  = (const int*)  d_in[0];
  const float* tok_emb = (const float*)d_in[1];
  const float* pos_emb = (const float*)d_in[2];
  const float* Wq = (const float*)d_in[3];  const float* bq = (const float*)d_in[4];
  const float* Wk = (const float*)d_in[5];  const float* bk = (const float*)d_in[6];
  const float* Wv = (const float*)d_in[7];  const float* bv = (const float*)d_in[8];
  const float* Wo = (const float*)d_in[9];  const float* bo = (const float*)d_in[10];
  const float* ln1w = (const float*)d_in[11]; const float* ln1b = (const float*)d_in[12];
  const float* ln2w = (const float*)d_in[13]; const float* ln2b = (const float*)d_in[14];
  const float* W1 = (const float*)d_in[15]; const float* b1 = (const float*)d_in[16];
  const float* W2 = (const float*)d_in[17]; const float* b2 = (const float*)d_in[18];
  const float* Wp = (const float*)d_in[19]; const float* bp = (const float*)d_in[20];
  const float* lnfw = (const float*)d_in[21]; const float* lnfb = (const float*)d_in[22];
  const float* Whead = (const float*)d_in[23]; const float* bhead = (const float*)d_in[24];
  float* out = (float*)d_out;

  unsigned char* wsb = (unsigned char*)d_ws;
  size_t cur = 0;
  auto carve = [&](size_t bytes) -> void* {
    void* p = wsb + cur;
    cur += (bytes + 255) & ~(size_t)255;
    return p;
  };

  // f16 weight caches (L2-resident working set ~179MB)
  _Float16* wq16 = (_Float16*)carve((size_t)L * D * D  * 2);
  _Float16* wk16 = (_Float16*)carve((size_t)L * D * HD * 2);
  _Float16* wv16 = (_Float16*)carve((size_t)L * D * HD * 2);
  _Float16* wo16 = (_Float16*)carve((size_t)L * D * D  * 2);
  _Float16* w116 = (_Float16*)carve((size_t)L * D * FF * 2);
  _Float16* w216 = (_Float16*)carve((size_t)L * D * FF * 2);
  _Float16* wp16 = (_Float16*)carve((size_t)L * FF * D * 2);
  _Float16* wh16 = (_Float16*)carve((size_t)D * V * 2);
  // activations
  float*    x    = (float*)   carve((size_t)M * D * 4);
  float*    x2   = (float*)   carve((size_t)M * D * 4);
  _Float16* h16  = (_Float16*)carve((size_t)M * D * 2);
  float*    qf   = (float*)   carve((size_t)M * D * 4);
  float*    kf   = (float*)   carve((size_t)M * HD * 4);
  float*    vf   = (float*)   carve((size_t)M * HD * 4);
  _Float16* att16= (_Float16*)carve((size_t)M * D * 2);
  float*    ubuf = (float*)   carve((size_t)M * FF * 4);
  float*    tbuf = (float*)   carve((size_t)M * FF * 4);
  _Float16* g16  = (_Float16*)carve((size_t)M * FF * 2);

  auto cvt = [&](const float* s, _Float16* d, size_t n) {
    unsigned g = (unsigned)((n + 255) / 256);
    k_cvt_f32_f16<<<g, 256, 0, stream>>>(s, d, n);
  };
  cvt(Wq, wq16, (size_t)L * D * D);
  cvt(Wk, wk16, (size_t)L * D * HD);
  cvt(Wv, wv16, (size_t)L * D * HD);
  cvt(Wo, wo16, (size_t)L * D * D);
  cvt(W1, w116, (size_t)L * D * FF);
  cvt(W2, w216, (size_t)L * D * FF);
  cvt(Wp, wp16, (size_t)L * FF * D);
  cvt(Whead, wh16, (size_t)D * V);

  auto gemm = [&](const _Float16* A, const _Float16* Bw, const float* bias,
                  const float* resid, float* C, int m, int n, int k) {
    dim3 g((n + BN - 1) / BN, m / BM);
    if (n % BN == 0)
      k_gemm_f16<true><<<g, 256, 0, stream>>>(A, Bw, bias, resid, C, m, n, k);
    else
      k_gemm_f16<false><<<g, 256, 0, stream>>>(A, Bw, bias, resid, C, m, n, k);
  };

  k_embed<<<(M * D + 255) / 256, 256, 0, stream>>>(tokens, tok_emb, pos_emb, x);

  for (int l = 0; l < L; ++l) {
    k_layernorm<<<M, 256, 0, stream>>>(x, ln1w + (size_t)l * D, ln1b + (size_t)l * D, h16, D);
    gemm(h16, wq16 + (size_t)l * D * D,  bq + (size_t)l * D,  nullptr, qf, M, D,  D);
    gemm(h16, wk16 + (size_t)l * D * HD, bk + (size_t)l * HD, nullptr, kf, M, HD, D);
    gemm(h16, wv16 + (size_t)l * D * HD, bv + (size_t)l * HD, nullptr, vf, M, HD, D);
    k_rope_q<<<(M * H * 16 + 255) / 256, 256, 0, stream>>>(qf);
    k_rope_k<<<(M * 16 + 255) / 256, 256, 0, stream>>>(kf);
    k_attn<<<B * H * T, 256, 0, stream>>>(qf, kf, vf, att16);
    gemm(att16, wo16 + (size_t)l * D * D, bo + (size_t)l * D, x, x2, M, D, D);
    k_layernorm<<<M, 256, 0, stream>>>(x2, ln2w + (size_t)l * D, ln2b + (size_t)l * D, h16, D);
    gemm(h16, w116 + (size_t)l * D * FF, b1 + (size_t)l * FF, nullptr, ubuf, M, FF, D);
    gemm(h16, w216 + (size_t)l * D * FF, b2 + (size_t)l * FF, nullptr, tbuf, M, FF, D);
    k_silu_gate<<<(M * FF + 255) / 256, 256, 0, stream>>>(ubuf, tbuf, g16, (size_t)M * FF);
    gemm(g16, wp16 + (size_t)l * FF * D, bp + (size_t)l * D, x2, x, M, D, FF);
  }

  k_layernorm<<<M, 256, 0, stream>>>(x, lnfw, lnfb, h16, D);
  gemm(h16, wh16, bhead, nullptr, out, M, V, D);
}
